// BiLSTM_CRF_78494822301882
// MI455X (gfx1250) — matrix operand order repaired and checked by reading, not compiled, not run on hardware
//
#include <hip/hip_runtime.h>
#include <hip/hip_bf16.h>
#include <stdint.h>

// ---------------------------------------------------------------------------
// Toolchain feature detection (TDM builtin exists on device pass only;
// arity differs: ROCm 7.2 = 5 args, amdgpu-toolchain (ships gfx1250_TDM.h) = 6)
// ---------------------------------------------------------------------------
#if defined(__has_builtin)
#  if __has_builtin(__builtin_amdgcn_tensor_load_to_lds)
#    define CDNA5_HAS_TDM 1
#  endif
#endif
#ifndef CDNA5_HAS_TDM
#  define CDNA5_HAS_TDM 0
#endif
#if defined(__has_include)
#  if __has_include(<hip/amd_detail/amd_gfx1250_TDM.h>)
#    define CDNA5_TDM_6ARG 1
#  endif
#endif
#ifndef CDNA5_TDM_6ARG
#  define CDNA5_TDM_6ARG 0
#endif

// ---------------------------------------------------------------------------
// Types for WMMA fragments (wave32, CDNA5)
// ---------------------------------------------------------------------------
typedef __attribute__((ext_vector_type(16))) __bf16       v16bf;
typedef __attribute__((ext_vector_type(8)))  float        v8f;
typedef __attribute__((ext_vector_type(4)))  unsigned int u32x4;
typedef __attribute__((ext_vector_type(4)))  int          i32x4;
typedef __attribute__((ext_vector_type(8)))  int          i32x8;

union BF16Frag { u32x4 u[2]; v16bf v; };

__device__ __forceinline__ unsigned short f2bf(float f) {
    unsigned u = __builtin_bit_cast(unsigned, f);
    u += 0x7FFFu + ((u >> 16) & 1u);          // round-to-nearest-even
    return (unsigned short)(u >> 16);
}
__device__ __forceinline__ float bf2f(unsigned short h) {
    unsigned u = ((unsigned)h) << 16;
    return __builtin_bit_cast(float, u);
}
__device__ __forceinline__ float sigf(float x)   { return 1.0f / (1.0f + __expf(-x)); }
__device__ __forceinline__ float tanh_f(float x) { return 2.0f / (1.0f + __expf(-2.0f * x)) - 1.0f; }

__device__ __forceinline__ v8f wmma_bf16(v16bf a, v16bf b, v8f c) {
    return __builtin_amdgcn_wmma_f32_16x16x32_bf16(
        /*neg_a=*/false, a, /*neg_b=*/false, b,
        /*c_mod=*/(short)0, c, /*reuse_a=*/false, /*reuse_b=*/false);
}

// ---------------------------------------------------------------------------
// Tensor Data Mover: async DMA of a 2D bf16 tile (rows x cols) global -> LDS.
// D# per CDNA5 ISA ch.8: group0 = {count/type, lds_addr, global_addr},
// group1 = {data_size, tensor dims, tile dims, dim0 stride}; groups 2/3 zero.
// ---------------------------------------------------------------------------
__device__ __forceinline__ void tdm_load_2d_bf16(
    unsigned lds_addr, const void* gptr,
    unsigned tensor_d0, unsigned tensor_d1,
    unsigned tile_d0, unsigned tile_d1, unsigned stride0_elems) {
#if CDNA5_HAS_TDM
    unsigned long long ga = (unsigned long long)(uintptr_t)gptr;
    u32x4 g0;
    g0[0] = 1u;                                              // count=1, user desc
    g0[1] = lds_addr;                                        // LDS byte address
    g0[2] = (unsigned)(ga & 0xFFFFFFFFu);                    // global_addr[31:0]
    g0[3] = (unsigned)((ga >> 32) & 0x1FFFFFFu) | (2u << 30); // addr[56:32] | type=2
    i32x8 g1;
    g1[0] = (int)(1u << 16);                                 // data_size=1 (2 bytes)
    g1[1] = (int)((tensor_d0 & 0xFFFFu) << 16);              // tensor_dim0[15:0]
    g1[2] = (int)(((tensor_d0 >> 16) & 0xFFFFu) |
                  ((tensor_d1 & 0xFFFFu) << 16));            // dim0[31:16] | dim1[15:0]
    g1[3] = (int)(((tensor_d1 >> 16) & 0xFFFFu) |
                  ((tile_d0 & 0xFFFFu) << 16));              // dim1[31:16] | tile_dim0
    g1[4] = (int)(tile_d1 & 0xFFFFu);                        // tile_dim1 (tile_dim2=0)
    g1[5] = (int)stride0_elems;                              // tensor_dim0_stride lo32
    g1[6] = 0;
    g1[7] = 0;
    i32x4 z4 = {0, 0, 0, 0};
#if CDNA5_TDM_6ARG
    i32x8 z8 = {0, 0, 0, 0, 0, 0, 0, 0};
    __builtin_amdgcn_tensor_load_to_lds(g0, g1, z4, z4, z8, 0);
#else
    __builtin_amdgcn_tensor_load_to_lds(g0, g1, z4, z4, 0);
#endif
#else
    (void)lds_addr; (void)gptr; (void)tensor_d0; (void)tensor_d1;
    (void)tile_d0; (void)tile_d1; (void)stride0_elems;
#endif
}

static constexpr int Bsz = 32, Ssz = 512, Hh = 256, Tt = 11;
static constexpr int G4H = 4 * Hh;       // 1024
static constexpr int H2  = 2 * Hh;       // 512

// ---------------------------------------------------------------------------
// f32 -> bf16 weight conversion (grid-stride)
// ---------------------------------------------------------------------------
__global__ void f32_to_bf16_kernel(const float* __restrict__ src,
                                   unsigned short* __restrict__ dst, int n) {
    for (int i = blockIdx.x * blockDim.x + threadIdx.x; i < n;
         i += gridDim.x * blockDim.x)
        dst[i] = f2bf(src[i]);
}

// ---------------------------------------------------------------------------
// Embedding gather + bf16 convert
// ---------------------------------------------------------------------------
__global__ void embed_kernel(const int* __restrict__ x,
                             const float* __restrict__ emb,
                             unsigned short* __restrict__ xe) {
    size_t i = (size_t)blockIdx.x * blockDim.x + threadIdx.x;   // over B*S*H
    if (i >= (size_t)Bsz * Ssz * Hh) return;
    int tok = (int)(i >> 8);
    int hh  = (int)(i & 255);
    int id  = x[tok];
    xe[i] = f2bf(emb[(size_t)id * Hh + hh]);
}

// ---------------------------------------------------------------------------
// Input-to-hidden GEMM: C[M x 1024] = A[M x K](bf16) * W[1024 x K]^T(bf16)+bias
// 256 threads = 8 waves, tile 64(M) x 128(N), K-step 32.
// Tiles staged by the Tensor Data Mover (double-buffered, TENSORcnt-tracked):
// wave 0 DMAs the A tile, wave 1 the B tile, while all waves compute on the
// other buffer. Fallback manual staging for toolchains without the builtin.
// ---------------------------------------------------------------------------
__global__ __launch_bounds__(256) void wih_gemm_kernel(
    const unsigned short* __restrict__ A, const unsigned short* __restrict__ W,
    const float* __restrict__ bias, float* __restrict__ C, int M, int N, int K) {
    __shared__ unsigned short sA[2][64 * 32];    // 2 x 4KB
    __shared__ unsigned short sB[2][128 * 32];   // 2 x 8KB

    const int tid  = threadIdx.x;
    const int lane = tid & 31, wid = tid >> 5;
    const int mBase = blockIdx.x * 64;
    const int nBase = blockIdx.y * 128;
    const int mw = (wid & 1) * 32;
    const int nw = (wid >> 1) * 32;
    const int nk = K / 32;

    v8f acc[2][2] = {};

#if CDNA5_HAS_TDM
    // prologue: kick off DMA of the first K-tile
    if (wid == 0)
        tdm_load_2d_bf16((unsigned)(uintptr_t)&sA[0][0], A + (size_t)mBase * K,
                         K, M, 32, 64, K);
    else if (wid == 1)
        tdm_load_2d_bf16((unsigned)(uintptr_t)&sB[0][0], W + (size_t)nBase * K,
                         K, N, 32, 128, K);
#endif

    for (int i = 0; i < nk; ++i) {
        const int p = i & 1;
#if CDNA5_HAS_TDM
        if (i + 1 < nk) {            // issue DMA of the next tile into buf p^1
            const int k0n = (i + 1) * 32;
            if (wid == 0)
                tdm_load_2d_bf16((unsigned)(uintptr_t)&sA[p ^ 1][0],
                                 A + (size_t)mBase * K + k0n, K, M, 32, 64, K);
            else if (wid == 1)
                tdm_load_2d_bf16((unsigned)(uintptr_t)&sB[p ^ 1][0],
                                 W + (size_t)nBase * K + k0n, K, N, 32, 128, K);
            if (wid < 2) __builtin_amdgcn_s_wait_tensorcnt((short)1);
        } else if (wid < 2) {
            __builtin_amdgcn_s_wait_tensorcnt((short)0);
        }
#else
        {   // manual staging fallback into buffer p
            int row = tid >> 2, ko = (tid & 3) * 8;
            int k0 = i * 32;
            u32x4 dA = *(const u32x4*)(A + (size_t)(mBase + row) * K + k0 + ko);
            *(u32x4*)(sA[p] + row * 32 + ko) = dA;
            #pragma unroll
            for (int q = 0; q < 2; ++q) {
                int rw = row + q * 64;
                u32x4 dB = *(const u32x4*)(W + (size_t)(nBase + rw) * K + k0 + ko);
                *(u32x4*)(sB[p] + rw * 32 + ko) = dB;
            }
        }
#endif
        __syncthreads();             // tile p visible to all waves

        v16bf aF[2], bF[2];
        #pragma unroll
        for (int mt = 0; mt < 2; ++mt) {     // A frag: lane<16 K0-7/16-23
            int r  = mw + mt * 16 + (lane & 15);
            int c0 = (lane >> 4) * 8;
            BF16Frag t;
            t.u[0] = *(const u32x4*)(sA[p] + r * 32 + c0);
            t.u[1] = *(const u32x4*)(sA[p] + r * 32 + c0 + 16);
            aF[mt] = t.v;
        }
        #pragma unroll
        for (int nt = 0; nt < 2; ++nt) {     // B frag: lane<16 K0-7/16-23
            int n  = nw + nt * 16 + (lane & 15);
            int kb = (lane >> 4) * 8;
            BF16Frag t;
            t.u[0] = *(const u32x4*)(sB[p] + n * 32 + kb);
            t.u[1] = *(const u32x4*)(sB[p] + n * 32 + kb + 16);
            bF[nt] = t.v;
        }
        #pragma unroll
        for (int nt = 0; nt < 2; ++nt)
            #pragma unroll
            for (int mt = 0; mt < 2; ++mt)
                acc[mt][nt] = wmma_bf16(aF[mt], bF[nt], acc[mt][nt]);
        __syncthreads();             // done with tile p before DMA reuses it
    }

    #pragma unroll
    for (int mt = 0; mt < 2; ++mt)
        #pragma unroll
        for (int nt = 0; nt < 2; ++nt)
            #pragma unroll
            for (int r = 0; r < 8; ++r) {
                int row = mBase + mw + mt * 16 + (lane >> 4) * 8 + r;
                int col = nBase + nw + nt * 16 + (lane & 15);
                C[(size_t)row * N + col] = acc[mt][nt][r] + bias[col];
            }
}

// ---------------------------------------------------------------------------
// LSTM recurrence, one direction per block (block 0 = fwd, 1 = bwd).
// 512 threads = 16 waves. Wave w owns hidden columns [w*16, w*16+16) in ALL
// FOUR gate blocks so i/f/g/o for a hidden unit live in the same lane ->
// cell state stays in registers for all 512 steps. h double-buffered in LDS.
// ---------------------------------------------------------------------------
__global__ __launch_bounds__(512) void lstm_dir_kernel(
    const float* __restrict__ xs_fwd, const float* __restrict__ xs_bwd,
    const unsigned short* __restrict__ whh_fwd,
    const unsigned short* __restrict__ whh_bwd,
    unsigned short* __restrict__ hout /* [B,S,2H] */) {
    __shared__ unsigned short hbuf[2][Bsz * Hh];   // 2 x 16KB

    const int dir = blockIdx.x;
    const float* xs = dir ? xs_bwd : xs_fwd;
    const unsigned short* whh = dir ? whh_bwd : whh_fwd;
    const int colOff = dir * Hh;

    const int tid = threadIdx.x, lane = tid & 31, wid = tid >> 5;

    for (int i = tid; i < Bsz * Hh; i += 512) hbuf[0][i] = 0;  // h0 = 0
    float creg[2][8] = {};                                      // c0 = 0
    __syncthreads();

    const int myCol = wid * 16 + (lane & 15);
    for (int s = 0; s < Ssz; ++s) {
        const int tt = dir ? (Ssz - 1 - s) : s;
        const int rb = s & 1, wb = rb ^ 1;

        if (s + 1 < Ssz) {           // global_prefetch for next step's xs
            int tn = dir ? (Ssz - 2 - s) : s + 1;
            #pragma unroll
            for (int mt = 0; mt < 2; ++mt) {
                int row = mt * 16 + (lane >> 4) * 8;
                __builtin_prefetch(&xs[((size_t)row * Ssz + tn) * G4H + myCol], 0, 1);
            }
        }

        // gates = h @ Whh^T : 2(M) x 4(gates) x 8(K) WMMAs per wave
        v8f acc[2][4] = {};
        #pragma unroll
        for (int kk = 0; kk < 8; ++kk) {
            v16bf aF[2];
            #pragma unroll
            for (int mt = 0; mt < 2; ++mt) {
                int r  = mt * 16 + (lane & 15);
                int c0 = kk * 32 + (lane >> 4) * 8;
                BF16Frag t;
                t.u[0] = *(const u32x4*)(hbuf[rb] + r * Hh + c0);
                t.u[1] = *(const u32x4*)(hbuf[rb] + r * Hh + c0 + 16);
                aF[mt] = t.v;
            }
            #pragma unroll
            for (int nt = 0; nt < 4; ++nt) {
                int n  = nt * Hh + myCol;
                int kb = kk * 32 + (lane >> 4) * 8;
                BF16Frag t;
                const unsigned short* ptr = whh + (size_t)n * Hh + kb;
                t.u[0] = *(const u32x4*)(ptr);
                t.u[1] = *(const u32x4*)(ptr + 16);
                v16bf bF = t.v;
                acc[0][nt] = wmma_bf16(aF[0], bF, acc[0][nt]);
                acc[1][nt] = wmma_bf16(aF[1], bF, acc[1][nt]);
            }
        }

        // cell update entirely in registers
        #pragma unroll
        for (int mt = 0; mt < 2; ++mt) {
            #pragma unroll
            for (int r = 0; r < 8; ++r) {
                int row = mt * 16 + (lane >> 4) * 8 + r;
                size_t xrow = ((size_t)row * Ssz + tt) * G4H;
                float iG = acc[mt][0][r] + xs[xrow + myCol];
                float fG = acc[mt][1][r] + xs[xrow + Hh + myCol];
                float gG = acc[mt][2][r] + xs[xrow + 2 * Hh + myCol];
                float oG = acc[mt][3][r] + xs[xrow + 3 * Hh + myCol];
                float c  = sigf(fG) * creg[mt][r] + sigf(iG) * tanh_f(gG);
                float h  = sigf(oG) * tanh_f(c);
                creg[mt][r] = c;
                unsigned short hb = f2bf(h);
                hbuf[wb][row * Hh + myCol] = hb;
                hout[((size_t)row * Ssz + tt) * H2 + colOff + myCol] = hb;
            }
        }
        __syncthreads();
    }
}

// ---------------------------------------------------------------------------
// Output projection: feats[bs][t] = h1[bs] . w_out[t] + b_out[t]   (T=11)
// ---------------------------------------------------------------------------
__global__ void outproj_kernel(const unsigned short* __restrict__ h1,
                               const float* __restrict__ w_out,
                               const float* __restrict__ b_out,
                               float* __restrict__ feats) {
    int i = blockIdx.x * blockDim.x + threadIdx.x;
    if (i >= Bsz * Ssz * Tt) return;
    int bs = i / Tt, t = i % Tt;
    const unsigned short* hr = h1 + (size_t)bs * H2;
    const float* wr = w_out + (size_t)t * H2;
    float acc = b_out[t];
    #pragma unroll 8
    for (int k = 0; k < H2; ++k) acc += bf2f(hr[k]) * wr[k];
    feats[(size_t)bs * Tt + t] = acc;
}

// ---------------------------------------------------------------------------
// Viterbi decode (T=11, B=32). One block; thread = (b, next_tag).
// ---------------------------------------------------------------------------
__global__ __launch_bounds__(384) void viterbi_kernel(
    const float* __restrict__ feats, const float* __restrict__ trans,
    int* __restrict__ bp, float* __restrict__ out_score,
    float* __restrict__ out_tags) {
    __shared__ float fv[2][Bsz][16];
    __shared__ float tr[Tt][16];

    const int tid = threadIdx.x;
    if (tid < Tt * Tt) tr[tid / Tt][tid % Tt] = trans[tid];
    const int b = tid / Tt, nt = tid % Tt;
    const bool act = tid < Bsz * Tt;
    if (act) fv[0][b][nt] = (nt == 9 /*START*/) ? 0.0f : -1000.0f;
    __syncthreads();

    for (int s = 0; s < Ssz; ++s) {
        const int rb = s & 1, wbuf = rb ^ 1;
        if (act) {
            float best = -3.4e38f; int bi = 0;
            #pragma unroll
            for (int p = 0; p < Tt; ++p) {
                float v = fv[rb][b][p] + tr[nt][p];
                if (v > best) { best = v; bi = p; }
            }
            fv[wbuf][b][nt] = best + feats[((size_t)b * Ssz + s) * Tt + nt];
            bp[(s * Bsz + b) * Tt + nt] = bi;
        }
        __syncthreads();
    }
    if (act && nt == 0) {
        float best = -3.4e38f; int last = 0;
        #pragma unroll
        for (int t = 0; t < Tt; ++t) {
            float v = fv[0][b][t] + tr[10 /*STOP*/][t];
            if (v > best) { best = v; last = t; }
        }
        out_score[b] = best;
        int tag = last;
        for (int s = Ssz - 1; s >= 0; --s) {
            out_tags[(size_t)b * Ssz + s] = (float)tag;
            tag = bp[(s * Bsz + b) * Tt + tag];
        }
    }
}

// ---------------------------------------------------------------------------
// Orchestration
// ---------------------------------------------------------------------------
extern "C" void kernel_launch(void* const* d_in, const int* in_sizes, int n_in,
                              void* d_out, int out_size, void* d_ws, size_t ws_size,
                              hipStream_t stream) {
    const int*   x     = (const int*)d_in[0];
    const float* emb   = (const float*)d_in[2];
    const float* wih0f = (const float*)d_in[3];
    const float* whh0f = (const float*)d_in[4];
    const float* b0f   = (const float*)d_in[5];
    const float* wih0b = (const float*)d_in[6];
    const float* whh0b = (const float*)d_in[7];
    const float* b0b   = (const float*)d_in[8];
    const float* wih1f = (const float*)d_in[9];
    const float* whh1f = (const float*)d_in[10];
    const float* b1f   = (const float*)d_in[11];
    const float* wih1b = (const float*)d_in[12];
    const float* whh1b = (const float*)d_in[13];
    const float* b1b   = (const float*)d_in[14];
    const float* w_out = (const float*)d_in[15];
    const float* b_out = (const float*)d_in[16];
    const float* trans = (const float*)d_in[17];

    unsigned short* wb      = (unsigned short*)d_ws;
    unsigned short* wih0f_h = wb;                 // 1024*256
    unsigned short* wih0b_h = wb + 262144;
    unsigned short* whh0f_h = wb + 524288;
    unsigned short* whh0b_h = wb + 786432;
    unsigned short* wih1f_h = wb + 1048576;       // 1024*512
    unsigned short* wih1b_h = wb + 1572864;
    unsigned short* whh1f_h = wb + 2097152;
    unsigned short* whh1b_h = wb + 2359296;
    unsigned short* xe      = wb + 2621440;       // B*S*H    = 4194304
    unsigned short* h0      = wb + 6815744;       // B*S*2H   = 8388608
    unsigned short* h1      = wb + 15204352;      // B*S*2H
    float* xs_f  = (float*)((char*)d_ws + 47185920ull);   // B*S*4H f32
    float* xs_b  = xs_f + 16777216;
    float* feats = xs_b + 16777216;               // B*S*T
    int*   bp    = (int*)(feats + 180224);        // S*B*T

    f32_to_bf16_kernel<<<256, 256, 0, stream>>>(wih0f, wih0f_h, 262144);
    f32_to_bf16_kernel<<<256, 256, 0, stream>>>(wih0b, wih0b_h, 262144);
    f32_to_bf16_kernel<<<256, 256, 0, stream>>>(whh0f, whh0f_h, 262144);
    f32_to_bf16_kernel<<<256, 256, 0, stream>>>(whh0b, whh0b_h, 262144);
    f32_to_bf16_kernel<<<256, 256, 0, stream>>>(wih1f, wih1f_h, 524288);
    f32_to_bf16_kernel<<<256, 256, 0, stream>>>(wih1b, wih1b_h, 524288);
    f32_to_bf16_kernel<<<256, 256, 0, stream>>>(whh1f, whh1f_h, 262144);
    f32_to_bf16_kernel<<<256, 256, 0, stream>>>(whh1b, whh1b_h, 262144);

    embed_kernel<<<16384, 256, 0, stream>>>(x, emb, xe);

    wih_gemm_kernel<<<dim3(256, 8), 256, 0, stream>>>(xe, wih0f_h, b0f, xs_f, 16384, 1024, 256);
    wih_gemm_kernel<<<dim3(256, 8), 256, 0, stream>>>(xe, wih0b_h, b0b, xs_b, 16384, 1024, 256);
    lstm_dir_kernel<<<2, 512, 0, stream>>>(xs_f, xs_b, whh0f_h, whh0b_h, h0);

    wih_gemm_kernel<<<dim3(256, 8), 256, 0, stream>>>(h0, wih1f_h, b1f, xs_f, 16384, 1024, 512);
    wih_gemm_kernel<<<dim3(256, 8), 256, 0, stream>>>(h0, wih1b_h, b1b, xs_b, 16384, 1024, 512);
    lstm_dir_kernel<<<2, 512, 0, stream>>>(xs_f, xs_b, whh1f_h, whh1b_h, h1);

    outproj_kernel<<<(Bsz * Ssz * Tt + 255) / 256, 256, 0, stream>>>(h1, w_out, b_out, feats);
    viterbi_kernel<<<1, 384, 0, stream>>>(feats, trans, bp,
                                          (float*)d_out, (float*)d_out + Bsz);
}
